// EdgeAttention_28518582846267
// MI455X (gfx1250) — matrix operand order repaired and checked by reading, not compiled, not run on hardware
//
#include <hip/hip_runtime.h>
#include <hip/hip_bf16.h>

typedef __attribute__((ext_vector_type(16))) __bf16 v16bf;
typedef __attribute__((ext_vector_type(8)))  float  v8f;

struct Q2 { uint4 a, b; };   // 32 bytes == one v16bf fragment

__device__ __forceinline__ unsigned short f2bf(float f) {
    unsigned u = __float_as_uint(f);
    return (unsigned short)((u + 0x7FFFu + ((u >> 16) & 1u)) >> 16);
}
__device__ __forceinline__ unsigned pack2(float x, float y) {
    return (unsigned)f2bf(x) | ((unsigned)f2bf(y) << 16);
}

// ---------------------------------------------------------------------------
// Tensor Data Mover support (guarded; 5-arg on clang-22/ROCm7.2, 6-arg on 23+)
// ---------------------------------------------------------------------------
#if defined(__gfx1250__) && __has_builtin(__builtin_amdgcn_tensor_load_to_lds) && \
    __has_builtin(__builtin_amdgcn_s_wait_tensorcnt)
#define HAVE_TDM 1
#else
#define HAVE_TDM 0
#endif

#if HAVE_TDM
typedef unsigned tdm_u32x4 __attribute__((ext_vector_type(4)));
typedef int      tdm_i32x8 __attribute__((ext_vector_type(8)));
typedef int      tdm_i32x4 __attribute__((ext_vector_type(4)));

// 1-D f32 tile: tensor_elems total, tile_elems per transfer, dest LDS offset.
__device__ __forceinline__ void tdm_load_1d(const void* gaddr, unsigned lds_byte_off,
                                            unsigned tile_elems,
                                            unsigned long long tensor_elems) {
    const unsigned long long ga = (unsigned long long)gaddr;
    tdm_u32x4 g0;
    g0.x = 0x1u;                                              // count=1
    g0.y = lds_byte_off;                                      // lds_addr
    g0.z = (unsigned)(ga & 0xFFFFFFFFu);                      // global_addr[31:0]
    g0.w = (unsigned)((ga >> 32) & 0x01FFFFFFu) | (2u << 30); // addr[56:32], type=2
    tdm_i32x8 g1;
    g1[0] = 0x20000;                                          // data_size=4B, mask=0
    g1[1] = (int)((tensor_elems & 0xFFFFu) << 16);            // tensor_dim0[15:0]
    g1[2] = (int)(((tensor_elems >> 16) & 0xFFFFu) | (1u << 16)); // dim0 hi | dim1 lo=1
    g1[3] = (int)((tile_elems & 0xFFFFu) << 16);              // dim1 hi=0 | tile_dim0
    g1[4] = 1;                                                // tile_dim1=1, tile_dim2=0
    g1[5] = (int)(tensor_elems & 0xFFFFFFFFu);                // stride0[31:0]
    g1[6] = (int)((tensor_elems >> 32) & 0xFFFFu);            // stride0[47:32]
    g1[7] = 0;
    tdm_i32x4 gz = {0, 0, 0, 0};
#if __clang_major__ >= 23
    tdm_i32x8 gz8 = {0, 0, 0, 0, 0, 0, 0, 0};
    __builtin_amdgcn_tensor_load_to_lds(g0, g1, gz, gz, gz8, 0);
#else
    __builtin_amdgcn_tensor_load_to_lds(g0, g1, gz, gz, 0);
#endif
}
__device__ __forceinline__ unsigned lds_off32(const void* p) {
    return (unsigned)(unsigned long long)p;   // LDS aperture: low 32 bits = offset
}
#endif

// ---------------------------------------------------------------------------
// Prep kernel: fused, transposed bf16 weights in workspace.
//   Wt[n][k], k<128 -> W1a[k][n]; 128..159 -> (W_edge@W1b)[k-128][n];
//   k>=160 -> W1c[k-160][n].  biasf[n] = b1[n] + (b_edge @ W1b)[n].
// ---------------------------------------------------------------------------
__global__ void prep_kernel(const float* __restrict__ W_edge,
                            const float* __restrict__ b_edge,
                            const float* __restrict__ W1,
                            const float* __restrict__ b1,
                            unsigned short* __restrict__ Wt,
                            float* __restrict__ biasf) {
    int n = threadIdx.x;
    if (n >= 128) return;
    unsigned short* wrow = Wt + (size_t)n * 288;
    #pragma unroll 4
    for (int k = 0; k < 128; ++k) wrow[k]       = f2bf(W1[(size_t)k * 128 + n]);
    #pragma unroll 4
    for (int k = 0; k < 128; ++k) wrow[160 + k] = f2bf(W1[(size_t)(256 + k) * 128 + n]);
    for (int k = 0; k < 32; ++k) {
        float s = 0.f;
        #pragma unroll 4
        for (int j = 0; j < 128; ++j)
            s += W_edge[(size_t)k * 128 + j] * W1[(size_t)(128 + j) * 128 + n];
        wrow[128 + k] = f2bf(s);
    }
    float bs = b1[n];
    #pragma unroll 4
    for (int j = 0; j < 128; ++j)
        bs += b_edge[j] * W1[(size_t)(128 + j) * 128 + n];
    biasf[n] = bs;
}

// ---------------------------------------------------------------------------
#define AROW   296
#define TILE_E 32

struct Stage { float4 s[4]; float4 d[4]; };

__device__ __forceinline__ void stage_load(const float* __restrict__ nodef,
                                           const long long* __restrict__ rows,
                                           const long long* __restrict__ cols,
                                           int nEdges, int e0, int m, int part,
                                           Stage& S) {
    int e = e0 + m; if (e >= nEdges) e = nEdges - 1;
    const long long r = rows[e];
    const long long c = cols[e];
    const float4* sp = (const float4*)(nodef + r * 128);
    const float4* dp = (const float4*)(nodef + c * 128);
    #pragma unroll
    for (int it = 0; it < 4; ++it) {
        S.s[it] = sp[part + it * 8];
        S.d[it] = dp[part + it * 8];
    }
}

__device__ __forceinline__ void stage_store(unsigned short* __restrict__ buf,
                                            int m, int part, const Stage& S,
                                            float4 ea) {
    unsigned short* rowp = buf + m * AROW;
    #pragma unroll
    for (int it = 0; it < 4; ++it) {
        const int f4 = part + it * 8;
        *(uint2*)&rowp[f4 * 4] =
            make_uint2(pack2(S.s[it].x, S.s[it].y), pack2(S.s[it].z, S.s[it].w));
        *(uint2*)&rowp[160 + f4 * 4] =
            make_uint2(pack2(S.d[it].x, S.d[it].y), pack2(S.d[it].z, S.d[it].w));
    }
    *(uint2*)&rowp[128 + part * 4] = make_uint2(pack2(ea.x, ea.y), pack2(ea.z, ea.w));
}

__device__ __forceinline__ Q2 lds_frag(const unsigned short* p) {
    Q2 q; q.a = *(const uint4*)p; q.b = *(const uint4*)(p + 8); return q;
}

__global__ void __launch_bounds__(256)
edge_attn_kernel(const float* __restrict__ nodef,
                 const float* __restrict__ eattr,
                 const long long* __restrict__ eidx,
                 const unsigned short* __restrict__ Wt,
                 const float* __restrict__ biasf,
                 const float* __restrict__ W2,
                 const float* __restrict__ b2,
                 float* __restrict__ out,
                 int nEdges, int nGroups) {
    __shared__ unsigned short Atile[2][TILE_E * AROW];   // 2 x 18,944 B
    __shared__ float partial[8 * TILE_E];
    __shared__ float EAraw[2][TILE_E * 32];              // TDM staging, 2 x 4 KB

    const int tid  = threadIdx.x;
    const int wave = tid >> 5;
    const int lane = tid & 31;
    const int lhi  = lane >> 4;
    const int llo  = lane & 15;
    const int m8   = tid >> 3;
    const int p8   = tid & 7;

    const int ncol = (wave << 4) + llo;
    v16bf wfrag[9];
    {
        const unsigned short* wb = Wt + (size_t)ncol * 288 + (lhi << 4);
        #pragma unroll
        for (int c = 0; c < 9; ++c) {
            const uint4* wp = (const uint4*)(wb + c * 32);
            Q2 q; q.a = wp[0]; q.b = wp[1];
            wfrag[c] = __builtin_bit_cast(v16bf, q);
        }
    }
    const float w2v   = W2[ncol];
    const float biasv = biasf[ncol];
    const float b2v   = b2[0];

    const long long* rows = eidx;
    const long long* cols = eidx + nEdges;
    const int stride = gridDim.x;
    const unsigned long long eaElems = (unsigned long long)nEdges * 32ull;

    // ---- prologue: DMA tile g0's edge_attr, stage first tile into buffer 0 ----
    const int g0idx = blockIdx.x;
#if HAVE_TDM
    if (wave == 0) {
        tdm_load_1d(eattr + (size_t)g0idx * TILE_E * 32, lds_off32(&EAraw[0][0]),
                    TILE_E * 32, eaElems);
        __builtin_amdgcn_s_wait_tensorcnt(0);
    }
#endif
    Stage S;
    stage_load(nodef, rows, cols, nEdges, g0idx * TILE_E, m8, p8, S);
    __syncthreads();   // publish EAraw[0]
    {
        float4 ea0;
#if HAVE_TDM
        ea0 = ((const float4*)&EAraw[0][m8 * 32])[p8];
#else
        int e = g0idx * TILE_E + m8; if (e >= nEdges) e = nEdges - 1;
        ea0 = ((const float4*)(eattr + (size_t)e * 32))[p8];
#endif
        stage_store(&Atile[0][0], m8, p8, S, ea0);
    }
    __syncthreads();

    int cur = 0;
    for (int g = g0idx; g < nGroups; g += stride) {
        const int e0 = g * TILE_E;
        const int gn = g + stride;
        const bool hasNext = gn < nGroups;

        // kick next tile's transfers: TDM for edge_attr, vector loads for gathers
#if HAVE_TDM
        if (hasNext && wave == 0)
            tdm_load_1d(eattr + (size_t)gn * TILE_E * 32,
                        lds_off32(&EAraw[cur ^ 1][0]), TILE_E * 32, eaElems);
#endif
        if (hasNext)
            stage_load(nodef, rows, cols, nEdges, gn * TILE_E, m8, p8, S);
        if (gn + stride < nGroups)
            __builtin_prefetch(eattr + (size_t)(gn + stride) * TILE_E * 32, 0, 0);

        // ---- 18 WMMAs: 2 independent chains, pipelined LDS fragment feeds ----
        const unsigned short* ab0 = &Atile[cur][llo * AROW + (lhi << 4)];
        const unsigned short* ab1 = ab0 + 16 * AROW;
        Q2 qa0 = lds_frag(ab0);
        Q2 qa1 = lds_frag(ab1);
        v8f acc0 = {}, acc1 = {};
        #pragma unroll
        for (int c = 0; c < 9; ++c) {
            v16bf a0 = __builtin_bit_cast(v16bf, qa0);
            v16bf a1 = __builtin_bit_cast(v16bf, qa1);
            if (c < 8) {
                qa0 = lds_frag(ab0 + (c + 1) * 32);
                qa1 = lds_frag(ab1 + (c + 1) * 32);
            }
            acc0 = __builtin_amdgcn_wmma_f32_16x16x32_bf16(
                false, a0, false, wfrag[c], (short)0, acc0, false, false);
            acc1 = __builtin_amdgcn_wmma_f32_16x16x32_bf16(
                false, a1, false, wfrag[c], (short)0, acc1, false, false);
        }

        // ---- epilogue: bias+relu, *W2 column, N-reduce across 16-lane halves ----
        float v0[8], v1[8];
        #pragma unroll
        for (int r = 0; r < 8; ++r) {
            float h0 = acc0[r] + biasv; h0 = h0 > 0.f ? h0 : 0.f;
            float h1 = acc1[r] + biasv; h1 = h1 > 0.f ? h1 : 0.f;
            v0[r] = h0 * w2v;
            v1[r] = h1 * w2v;
        }
        #pragma unroll
        for (int msk = 1; msk < 16; msk <<= 1) {
            #pragma unroll
            for (int r = 0; r < 8; ++r) {
                v0[r] += __shfl_xor(v0[r], msk, 32);
                v1[r] += __shfl_xor(v1[r], msk, 32);
            }
        }
        if (llo == 0) {
            #pragma unroll
            for (int r = 0; r < 8; ++r) {
                partial[wave * TILE_E + lhi * 8 + r]      = v0[r];
                partial[wave * TILE_E + 16 + lhi * 8 + r] = v1[r];
            }
        }
#if HAVE_TDM
        if (wave == 0) __builtin_amdgcn_s_wait_tensorcnt(0);  // EAraw[cur^1] ready
#endif
        __syncthreads();   // publishes partial[] and EAraw[cur^1]

        if (tid < TILE_E && (e0 + tid) < nEdges) {
            float s = b2v;
            #pragma unroll
            for (int w = 0; w < 8; ++w) s += partial[w * TILE_E + tid];
            out[e0 + tid] = 1.f / (1.f + __expf(-s));
        }

        if (hasNext) {
            float4 ean;
#if HAVE_TDM
            ean = ((const float4*)&EAraw[cur ^ 1][m8 * 32])[p8];
#else
            int e = gn * TILE_E + m8; if (e >= nEdges) e = nEdges - 1;
            ean = ((const float4*)(eattr + (size_t)e * 32))[p8];
#endif
            stage_store(&Atile[cur ^ 1][0], m8, p8, S, ean);
        }
        __syncthreads();
        cur ^= 1;
    }
}

// ---------------------------------------------------------------------------
extern "C" void kernel_launch(void* const* d_in, const int* in_sizes, int n_in,
                              void* d_out, int out_size, void* d_ws, size_t ws_size,
                              hipStream_t stream) {
    const float*     nodef  = (const float*)d_in[0];
    const float*     eattr  = (const float*)d_in[1];
    const float*     W_edge = (const float*)d_in[2];
    const float*     b_edge = (const float*)d_in[3];
    const float*     W1     = (const float*)d_in[4];
    const float*     b1     = (const float*)d_in[5];
    const float*     W2     = (const float*)d_in[6];
    const float*     b2     = (const float*)d_in[7];
    const long long* eidx   = (const long long*)d_in[8];

    const int nEdges = in_sizes[1] / 32;

    unsigned short* Wt    = (unsigned short*)d_ws;
    float*          biasf = (float*)((char*)d_ws + 128 * 288 * sizeof(unsigned short));

    prep_kernel<<<1, 128, 0, stream>>>(W_edge, b_edge, W1, b1, Wt, biasf);

    const int nGroups = (nEdges + TILE_E - 1) / TILE_E;
    const int grid    = nGroups < 4096 ? nGroups : 4096;
    edge_attn_kernel<<<grid, 256, 0, stream>>>(nodef, eattr, eidx, Wt, biasf,
                                               W2, b2, (float*)d_out,
                                               nEdges, nGroups);
}
